// GAT_11123965297098
// MI455X (gfx1250) — matrix operand order repaired and checked by reading, not compiled, not run on hardware
//
#include <hip/hip_runtime.h>
#include <math.h>

// Problem sizes (fixed by the reference)
#define NN 4096
#define MM 8192
#define LRELU_A  0.2f
#define MASK_NEG -9e15f
#define BN_SC    0.9999950000374997f   // 1/sqrt(1+1e-5)

typedef __attribute__((ext_vector_type(16))) __bf16 v16bf;
typedef __attribute__((ext_vector_type(8)))  float  v8f;

// ---------------- cross-lane helpers (wave32) ----------------
__device__ __forceinline__ float lane_bcast(float v, int srcLane) {
  return __builtin_bit_cast(float,
      __builtin_amdgcn_ds_bpermute(srcLane << 2, __builtin_bit_cast(int, v)));
}
__device__ __forceinline__ float lane_xor16(float v) {
  int src = (((int)threadIdx.x) & 31) ^ 16;
  return __builtin_bit_cast(float,
      __builtin_amdgcn_ds_bpermute(src << 2, __builtin_bit_cast(int, v)));
}

// =============================================================
// Fused streaming kernel:
//   SOFTMAX=1 : out[16-row block, F] = softmax_row(masked lrelu(s+d)) @ hd
//   SOFTMAX=0 : out = mask @ hd              (dense-mask pooling / SpMM)
// One wave handles NT 16-col tiles; blockDim.y waves split columns and
// share the same mask rows through the WGP cache (mask read once).
// A-fragment (weights) built in-register, contiguous k order (k = 16*kh + s);
// B-fragment = contiguous 16 bf16 from hdT (transposed hd).
// =============================================================
template<int NT, bool SOFTMAX>
__global__ __launch_bounds__(128) void k_stream(
    float* __restrict__ out, const float* __restrict__ mask,
    const __bf16* __restrict__ hdT, const float* __restrict__ sv_,
    const float* __restrict__ dv_, int Q, int F)
{
  const int lane = threadIdx.x;     // 0..31
  const int r    = lane & 15;       // A-layout row within tile
  const int kh   = lane >> 4;       // K half
  const int prow = blockIdx.x * 16 + r;
  const int colbase = ((int)(blockIdx.y * blockDim.y + threadIdx.y)) * (NT * 16);
  const float* mrow = mask + (size_t)prow * Q;

  float sval = SOFTMAX ? sv_[prow] : 0.0f;
  float m    = -INFINITY;           // running row max (A-layout lanes)
  float lsum = 0.0f;                // running row sum-of-exp (per half-lane)
  v8f acc[NT] = {};

  for (int k0 = 0; k0 < Q; k0 += 32) {
    if (k0 + 256 < Q) __builtin_prefetch(mrow + k0 + 256, 0, 1);  // global_prefetch_b8

    float w[16];
    float tmax = -INFINITY;
#pragma unroll
    for (int v = 0; v < 8; ++v) {
      // A-layout K indices (contiguous, matches B): half0 -> 0..15 ; half1 -> 16..31
      const int k = k0 + 2 * v + kh * 16;
      const float2 mk = *(const float2*)(mrow + k);
      if (SOFTMAX) {
        const float2 dk = *(const float2*)(dv_ + k);
        float l0 = sval + dk.x, l1 = sval + dk.y;
        l0 = (l0 > 0.0f) ? l0 : LRELU_A * l0;
        l1 = (l1 > 0.0f) ? l1 : LRELU_A * l1;
        l0 = (mk.x > 0.0f) ? l0 : MASK_NEG;
        l1 = (mk.y > 0.0f) ? l1 : MASK_NEG;
        w[2 * v] = l0; w[2 * v + 1] = l1;
        tmax = fmaxf(tmax, fmaxf(l0, l1));
      } else {
        w[2 * v] = mk.x; w[2 * v + 1] = mk.y;
      }
    }

    if (SOFTMAX) {
      // combine the two half-lanes so both agree on the per-row tile max
      tmax = fmaxf(tmax, lane_xor16(tmax));
      const float mnew  = fmaxf(m, tmax);
      const float alpha = __expf(m - mnew);   // 0 on first tile (m = -inf)
      m = mnew;
      lsum *= alpha;
      // rescale accumulators: C-fragment row of VGPR j is (j + 8*kh)
      float arow[8];
#pragma unroll
      for (int j = 0; j < 8; ++j) arow[j] = lane_bcast(alpha, j + 8 * kh);
#pragma unroll
      for (int t = 0; t < NT; ++t)
#pragma unroll
        for (int j = 0; j < 8; ++j) acc[t][j] *= arow[j];
#pragma unroll
      for (int i = 0; i < 16; ++i) { w[i] = __expf(w[i] - m); lsum += w[i]; }
    }

    v16bf afrag;
#pragma unroll
    for (int i = 0; i < 16; ++i) afrag[i] = (__bf16)w[i];

#pragma unroll
    for (int t = 0; t < NT; ++t) {
      const int n = colbase + t * 16 + r;     // B-layout: lane = column
      const v16bf bfrag = *(const v16bf*)(hdT + (size_t)n * Q + k0 + kh * 16);
      acc[t] = __builtin_amdgcn_wmma_f32_16x16x32_bf16(
          false, afrag, false, bfrag, (short)0, acc[t], false, false);
    }
  }

  float linv[8];
  if (SOFTMAX) {
    lsum += lane_xor16(lsum);                  // full row sum on both halves
#pragma unroll
    for (int j = 0; j < 8; ++j) linv[j] = 1.0f / lane_bcast(lsum, j + 8 * kh);
  } else {
#pragma unroll
    for (int j = 0; j < 8; ++j) linv[j] = 1.0f;
  }
#pragma unroll
  for (int t = 0; t < NT; ++t) {
    const int col = colbase + t * 16 + r;
#pragma unroll
    for (int j = 0; j < 8; ++j) {
      const int row = blockIdx.x * 16 + j + 8 * kh;
      out[(size_t)row * F + col] = acc[t][j] * linv[j];
    }
  }
}

// =============================================================
// Small dense GEMM:  out[P,N] = act((A[P,K] @ B) + bias) * scale
// B given pre-transposed as bf16 Bt[N,K]; one wave per 16x(NT*16) tile.
// =============================================================
template<int NT>
__global__ __launch_bounds__(32) void k_gemm(
    float* __restrict__ out, const float* __restrict__ A,
    const __bf16* __restrict__ Bt, const float* __restrict__ bias,
    float scale, int relu, int K, int N)
{
  const int lane = threadIdx.x;
  const int r = lane & 15, kh = lane >> 4;
  const int prow = blockIdx.x * 16 + r;
  const int colbase = blockIdx.y * (NT * 16);
  const float* arow = A + (size_t)prow * K;
  v8f acc[NT] = {};
  for (int k0 = 0; k0 < K; k0 += 32) {
    v16bf af;
#pragma unroll
    for (int v = 0; v < 8; ++v) {
      const int k = k0 + 2 * v + kh * 16;
      const float2 a2 = *(const float2*)(arow + k);
      af[2 * v] = (__bf16)a2.x; af[2 * v + 1] = (__bf16)a2.y;
    }
#pragma unroll
    for (int t = 0; t < NT; ++t) {
      const int n = colbase + t * 16 + r;
      const v16bf bf_ = *(const v16bf*)(Bt + (size_t)n * K + k0 + kh * 16);
      acc[t] = __builtin_amdgcn_wmma_f32_16x16x32_bf16(
          false, af, false, bf_, (short)0, acc[t], false, false);
    }
  }
#pragma unroll
  for (int t = 0; t < NT; ++t) {
    const int col = colbase + t * 16 + r;
    const float b = bias ? bias[col] : 0.0f;
#pragma unroll
    for (int j = 0; j < 8; ++j) {
      const int row = blockIdx.x * 16 + j + 8 * kh;
      float v = (acc[t][j] + b) * scale;
      if (relu) v = fmaxf(v, 0.0f);
      out[(size_t)row * N + col] = v;
    }
  }
}

// ---------------- LDS-tiled transpose (f32 -> f32 or bf16) ----------------
template<typename OutT>
__global__ __launch_bounds__(256) void k_transpose(
    OutT* __restrict__ dst, const float* __restrict__ src, int R, int C)
{
  __shared__ float tile[32][33];
  const int c0 = blockIdx.x * 32, r0 = blockIdx.y * 32;
  const int tx = threadIdx.x, ty = threadIdx.y;   // 32 x 8
#pragma unroll
  for (int i = 0; i < 32; i += 8) {
    int rr = r0 + ty + i, cc = c0 + tx;
    tile[ty + i][tx] = (rr < R && cc < C) ? src[(size_t)rr * C + cc] : 0.0f;
  }
  __syncthreads();
#pragma unroll
  for (int i = 0; i < 32; i += 8) {
    int cc = c0 + ty + i, rr = r0 + tx;
    if (cc < C && rr < R) dst[(size_t)cc * R + rr] = (OutT)tile[tx][ty + i];
  }
}

// ---------------- small elementwise kernels ----------------
__global__ void k_rowdot(float* __restrict__ dst, const float* __restrict__ H,
                         const float* __restrict__ a, int F)
{
  int p = blockIdx.x * blockDim.x + threadIdx.x;
  const float* hp = H + (size_t)p * F;
  float acc = 0.0f;
  for (int f = 0; f < F; ++f) acc += hp[f] * a[f];
  dst[p] = acc;
}

__global__ void k_elu_concat(float* __restrict__ dst, const float* __restrict__ x,
                             const float* __restrict__ y, int F)
{
  int idx = blockIdx.x * blockDim.x + threadIdx.x;   // over P*F, exact
  int p = idx / F, f = idx % F;
  float a = x[idx], b = y[idx];
  a = (a > 0.0f) ? a : (__expf(a) - 1.0f);
  b = (b > 0.0f) ? b : (__expf(b) - 1.0f);
  dst[(size_t)p * (2 * F) + f] = a;
  dst[(size_t)p * (2 * F) + F + f] = b;
}

__global__ void k_add(float* __restrict__ dst, const float* __restrict__ x,
                      const float* __restrict__ y)
{
  int i = blockIdx.x * blockDim.x + threadIdx.x;
  dst[i] = x[i] + y[i];
}

__global__ void k_elu_logsoftmax16(float* __restrict__ dst, const float* __restrict__ src)
{
  int p = blockIdx.x * blockDim.x + threadIdx.x;
  float v[16]; float mx = -INFINITY;
#pragma unroll
  for (int i = 0; i < 16; ++i) {
    float t = src[(size_t)p * 16 + i];
    t = (t > 0.0f) ? t : (__expf(t) - 1.0f);
    v[i] = t; mx = fmaxf(mx, t);
  }
  float s = 0.0f;
#pragma unroll
  for (int i = 0; i < 16; ++i) s += __expf(v[i] - mx);
  float lse = mx + logf(s);
#pragma unroll
  for (int i = 0; i < 16; ++i) dst[(size_t)p * 16 + i] = v[i] - lse;
}

// ---------------- host helpers ----------------
static inline void tr_bf16(__bf16* dst, const float* src, int R, int C, hipStream_t st) {
  dim3 g((C + 31) / 32, (R + 31) / 32), b(32, 8);
  k_transpose<__bf16><<<g, b, 0, st>>>(dst, src, R, C);
}
static inline void tr_f32(float* dst, const float* src, int R, int C, hipStream_t st) {
  dim3 g((C + 31) / 32, (R + 31) / 32), b(32, 8);
  k_transpose<float><<<g, b, 0, st>>>(dst, src, R, C);
}
static inline void gemm_host(float* out, const float* A, const float* Bf32, __bf16* Wt,
                             const float* bias, float scale, int relu,
                             int P, int K, int N, hipStream_t st) {
  tr_bf16(Wt, Bf32, K, N, st);                       // Bt[N,K] bf16
  if (N % 64 == 0) { dim3 g(P / 16, N / 64), b(32);  k_gemm<4><<<g, b, 0, st>>>(out, A, Wt, bias, scale, relu, K, N); }
  else             { dim3 g(P / 16, N / 16), b(32);  k_gemm<1><<<g, b, 0, st>>>(out, A, Wt, bias, scale, relu, K, N); }
}
static inline void attn_host(float* out, const float* mask, const __bf16* hdT,
                             const float* sv, const float* dv, int P, int Q, int F, hipStream_t st) {
  if (F == 16) { dim3 g(P / 16, 1), b(32, 1);     k_stream<1, true><<<g, b, 0, st>>>(out, mask, hdT, sv, dv, Q, F); }
  else         { dim3 g(P / 16, 1), b(32, F / 64); k_stream<4, true><<<g, b, 0, st>>>(out, mask, hdT, sv, dv, Q, F); }
}
static inline void spmm_host(float* out, const float* mask, const __bf16* hT,
                             int P, int F, hipStream_t st) {
  dim3 g(P / 16, 1), b(32, F / 64);
  k_stream<4, false><<<g, b, 0, st>>>(out, mask, hT, nullptr, nullptr, P, F);
}

// =============================================================
extern "C" void kernel_launch(void* const* d_in, const int* in_sizes, int n_in,
                              void* d_out, int out_size, void* d_ws, size_t ws_size,
                              hipStream_t stream) {
  (void)in_sizes; (void)n_in; (void)out_size; (void)ws_size;
  const float* x    = (const float*)d_in[0];
  const float* ex   = (const float*)d_in[1];
  const float* adj  = (const float*)d_in[2];
  const float* eadj = (const float*)d_in[3];
  const float* nem  = (const float*)d_in[4];
  const float* L[3][6];
  for (int li = 0; li < 3; ++li)
    for (int k = 0; k < 6; ++k) L[li][k] = (const float*)d_in[5 + li * 6 + k];
  const float* m0_W1 = (const float*)d_in[23]; const float* m0_b1 = (const float*)d_in[24];
  const float* m0_W2 = (const float*)d_in[25]; const float* m0_b2 = (const float*)d_in[26];
  const float* m1_W1 = (const float*)d_in[27]; const float* m1_b1 = (const float*)d_in[28];
  const float* m1_W2 = (const float*)d_in[29]; const float* m1_b2 = (const float*)d_in[30];
  float* outp = (float*)d_out;

  // ---- workspace carve ----
  char* wsb = (char*)d_ws; size_t off = 0;
  auto alloc = [&](size_t bytes) -> void* {
    off = (off + 255) & ~(size_t)255; void* p = wsb + off; off += bytes; return p;
  };
  float*  nemT  = (float*)alloc((size_t)MM * NN * 4);   // n_e_adj^T for Gn
  float*  bufHa = (float*)alloc((size_t)NN * 256 * 4);
  float*  bufGa = (float*)alloc((size_t)MM * 256 * 4);
  float*  bufHb = (float*)alloc((size_t)NN * 256 * 4);
  float*  bufGb = (float*)alloc((size_t)MM * 256 * 4);
  float*  hbuf  = (float*)alloc((size_t)NN * 128 * 4);
  float*  gbuf  = (float*)alloc((size_t)MM * 128 * 4);
  float*  Hn    = (float*)alloc((size_t)NN * 128 * 4);
  float*  He    = (float*)alloc((size_t)NN * 128 * 4);
  float*  Ge    = (float*)alloc((size_t)MM * 128 * 4);
  float*  Gn    = (float*)alloc((size_t)MM * 128 * 4);
  __bf16* hT    = (__bf16*)alloc((size_t)128 * NN * 2);
  __bf16* gT    = (__bf16*)alloc((size_t)128 * MM * 2);
  __bf16* HT    = (__bf16*)alloc((size_t)256 * MM * 2);
  __bf16* Wt    = (__bf16*)alloc((size_t)256 * 256 * 2);
  float*  svec  = (float*)alloc((size_t)MM * 4);
  float*  dvec  = (float*)alloc((size_t)MM * 4);
  float*  pooled= (float*)alloc((size_t)MM * 256 * 4);
  float*  t1    = (float*)alloc((size_t)MM * 64 * 4);
  float*  hn_a  = (float*)alloc((size_t)NN * 64 * 4);
  float*  hn_b  = (float*)alloc((size_t)NN * 64 * 4);
  float*  he_a  = (float*)alloc((size_t)MM * 64 * 4);
  float*  he_b  = (float*)alloc((size_t)MM * 64 * 4);
  float*  fn    = (float*)alloc((size_t)NN * 16 * 4);
  float*  fe    = (float*)alloc((size_t)MM * 16 * 4);

  tr_f32(nemT, nem, NN, MM, stream);   // [MM, NN]

  auto gat_layer = [&](const float* xh, int KN, const float* xg, int KE,
                       const float* const* Lw, int F, bool concat,
                       float* outH, float* outG) {
    const float* Wn = Lw[0], *We = Lw[1];
    const float* ann = Lw[2], *ane = Lw[3], *aee = Lw[4], *aen = Lw[5];
    gemm_host(hbuf, xh, Wn, Wt, nullptr, 1.0f, 0, NN, KN, F, stream);
    gemm_host(gbuf, xg, We, Wt, nullptr, 1.0f, 0, MM, KE, F, stream);
    tr_bf16(hT, hbuf, NN, F, stream);   // [F, NN]
    tr_bf16(gT, gbuf, MM, F, stream);   // [F, MM]
    // Hn = att(h, h, ann, adj)
    k_rowdot<<<NN / 256, 256, 0, stream>>>(svec, hbuf, ann, F);
    k_rowdot<<<NN / 256, 256, 0, stream>>>(dvec, hbuf, ann + F, F);
    attn_host(Hn, adj, hT, svec, dvec, NN, NN, F, stream);
    // He = att(h, g, ane, nem)
    k_rowdot<<<NN / 256, 256, 0, stream>>>(svec, hbuf, ane, F);
    k_rowdot<<<MM / 256, 256, 0, stream>>>(dvec, gbuf, ane + F, F);
    attn_host(He, nem, gT, svec, dvec, NN, MM, F, stream);
    // Ge = att(g, g, aee, eadj)
    k_rowdot<<<MM / 256, 256, 0, stream>>>(svec, gbuf, aee, F);
    k_rowdot<<<MM / 256, 256, 0, stream>>>(dvec, gbuf, aee + F, F);
    attn_host(Ge, eadj, gT, svec, dvec, MM, MM, F, stream);
    // Gn = att(g, h, aen, nem^T)
    k_rowdot<<<MM / 256, 256, 0, stream>>>(svec, gbuf, aen, F);
    k_rowdot<<<NN / 256, 256, 0, stream>>>(dvec, hbuf, aen + F, F);
    attn_host(Gn, nemT, hT, svec, dvec, MM, NN, F, stream);
    if (concat) {
      k_elu_concat<<<(NN * F) / 256, 256, 0, stream>>>(outH, Hn, He, F);
      k_elu_concat<<<(MM * F) / 256, 256, 0, stream>>>(outG, Ge, Gn, F);
    } else {
      k_add<<<(NN * F) / 256, 256, 0, stream>>>(outH, Hn, He);
      k_add<<<(MM * F) / 256, 256, 0, stream>>>(outG, Ge, Gn);
    }
  };

  auto next_layer = [&](float* out, const float* hin, int P, int Fin, const float* Amask,
                        const float* W1, const float* b1, const float* W2, const float* b2) {
    tr_bf16(HT, hin, P, Fin, stream);              // [Fin, P]
    spmm_host(pooled, Amask, HT, P, Fin, stream);  // pooled = Amask @ hin
    gemm_host(t1, pooled, W1, Wt, b1, BN_SC, 1, P, Fin, 64, stream);
    gemm_host(out, t1, W2, Wt, b2, BN_SC, 1, P, 64, 64, stream);
  };

  // Layer 0: (128, 64) -> 64, concat -> 128 features
  gat_layer(x, 128, ex, 64, L[0], 64, true, bufHa, bufGa);
  // Layer 1: (128, 128) -> 128, concat -> 256 features
  gat_layer(bufHa, 128, bufGa, 128, L[1], 128, true, bufHb, bufGb);
  // Pool + MLP stacks (nodes, then edges)
  next_layer(hn_a, bufHb, NN, 256, adj,  m0_W1, m0_b1, m0_W2, m0_b2);
  next_layer(hn_b, hn_a,  NN,  64, adj,  m1_W1, m1_b1, m1_W2, m1_b2);
  next_layer(he_a, bufGb, MM, 256, eadj, m0_W1, m0_b1, m0_W2, m0_b2);
  next_layer(he_b, he_a,  MM,  64, eadj, m1_W1, m1_b1, m1_W2, m1_b2);
  // Layer 2: (64, 64) -> 16, no concat
  gat_layer(hn_b, 64, he_b, 64, L[2], 16, false, fn, fe);
  // elu + log_softmax into d_out (nodes first, then edges)
  k_elu_logsoftmax16<<<NN / 256, 256, 0, stream>>>(outp, fn);
  k_elu_logsoftmax16<<<MM / 256, 256, 0, stream>>>(outp + (size_t)NN * 16, fe);
}
